// FieldDecoder_15874199126284
// MI455X (gfx1250) — matrix operand order repaired and checked by reading, not compiled, not run on hardware
//
#include <hip/hip_runtime.h>

// ---------------------------------------------------------------------------
// FieldDecoder for MI455X (gfx1250, wave32, WMMA)
//   proj = einsum('bnd,xyd->bnxy')  -> fused atomic scatter-sum -> analytic mean
// GEMM precision: fp32 emulated via split-f16 (hi+lo), f32 accumulate:
//   A*B ~= Alo*Bhi + Ahi*Blo + Ahi*Bhi       (3x v_wmma_f32_16x16x32_f16)
// WG tile 64x256, wave tile 32x64 (8 accumulators), ping-pong LDS A staging.
// ---------------------------------------------------------------------------

typedef __attribute__((ext_vector_type(16))) _Float16 v16h;
typedef __attribute__((ext_vector_type(8)))  _Float16 v8h;
typedef __attribute__((ext_vector_type(4)))  _Float16 v4h;
typedef __attribute__((ext_vector_type(8)))  float    v8f;
typedef __attribute__((ext_vector_type(4)))  float    v4f;

#define B_     16
#define N_TOK  4096      // NH*NW
#define DO_    768
#define KHW    256       // KH*KW
#define H_     512
#define W_     512
#define E_     (H_ * W_)  // 262144

#define BM     64        // tokens (GEMM M) per workgroup
#define KT     32        // K per step (f16 WMMA depth)
#define NKT    (DO_ / KT)
#define LDA    40        // LDS row stride in halves (32 + 8 pad)
#define PL     (BM * LDA)     // plane stride (hi -> lo) in halves
#define BUFS   (2 * PL)       // ping-pong buffer stride in halves

// ------------------------------ init / prep --------------------------------

__global__ void fd_zero_out(float* __restrict__ out) {
  size_t i = (size_t)blockIdx.x * blockDim.x + threadIdx.x;
  v4f z = {0.f, 0.f, 0.f, 0.f};
  ((v4f*)out)[i] = z;
}

__global__ void fd_weight_split(const float* __restrict__ w,
                                _Float16* __restrict__ whi,
                                _Float16* __restrict__ wlo) {
  int i = blockIdx.x * blockDim.x + threadIdx.x;   // < 256*768
  float x = w[i];
  _Float16 h = (_Float16)x;
  whi[i] = h;
  wlo[i] = (_Float16)(x - (float)h);
}

// --------------------------- GEMM + scatter-sum ----------------------------

__global__ __launch_bounds__(256)
void fd_gemm_scatter(const float* __restrict__ tgt,
                     const _Float16* __restrict__ whi,
                     const _Float16* __restrict__ wlo,
                     const int* __restrict__ idx,
                     float* __restrict__ out) {
  // [buf][plane hi/lo][row 64][col 32 (+8 pad)]
  __shared__ _Float16 As[2 * BUFS];

  const int b    = blockIdx.y;
  const int n0   = blockIdx.x * BM;
  const int tid  = threadIdx.x;
  const int lane = tid & 31;
  const int wid  = tid >> 5;                 // 8 waves
  const int waveM = wid >> 2;                // 0..1 -> 32-row half of WG tile
  const int waveN = wid & 3;                 // 0..3 -> 64-col subtile
  const int lrow  = lane & 15;
  const int lhalf = lane >> 4;               // 0/1

  const float* tgt_b = tgt + (size_t)b * N_TOK * DO_;

  // cooperative A staging: 256 threads x 8 floats = 64x32 tile
  const int arow = tid >> 2;                 // 0..63
  const int acol = (tid & 3) * 8;            // 0,8,16,24
  const float* arow_p = tgt_b + (size_t)(n0 + arow) * DO_ + acol;

  v8f acc[2][4] = {};

  // B fragment lane base (contiguous k order: lane = column, lanes 16..31
  // hold K=16..31) — 32B contiguous v16h loads from the split-f16 weight.
  const int colb = waveN * 64 + lrow;
  const _Float16* whi_l = whi + (size_t)colb * DO_ + lhalf * 16;
  const _Float16* wlo_l = wlo + (size_t)colb * DO_ + lhalf * 16;

  // ---- split one 8-float strip into hi/lo and store to LDS buffer ----
  auto store_strip = [&](int buf, v4f v0, v4f v1) {
    _Float16* dh = &As[buf * BUFS + arow * LDA + acol];
    _Float16* dl = dh + PL;
    _Float16 h0 = (_Float16)v0.x, h1 = (_Float16)v0.y,
             h2 = (_Float16)v0.z, h3 = (_Float16)v0.w;
    _Float16 h4 = (_Float16)v1.x, h5 = (_Float16)v1.y,
             h6 = (_Float16)v1.z, h7 = (_Float16)v1.w;
    v4h hv0 = {h0, h1, h2, h3};
    v4h hv1 = {h4, h5, h6, h7};
    v4h lv0 = {(_Float16)(v0.x - (float)h0), (_Float16)(v0.y - (float)h1),
               (_Float16)(v0.z - (float)h2), (_Float16)(v0.w - (float)h3)};
    v4h lv1 = {(_Float16)(v1.x - (float)h4), (_Float16)(v1.y - (float)h5),
               (_Float16)(v1.z - (float)h6), (_Float16)(v1.w - (float)h7)};
    *(v4h*)(dh)     = hv0;
    *(v4h*)(dh + 4) = hv1;
    *(v4h*)(dl)     = lv0;
    *(v4h*)(dl + 4) = lv1;
  };

  // ---- prologue: stage k-tile 0 into buffer 0 ----
  {
    v4f v0 = *(const v4f*)(arow_p);
    v4f v1 = *(const v4f*)(arow_p + 4);
    store_strip(0, v0, v1);
  }
  __syncthreads();

  int buf = 0;
  for (int kt = 0; kt < NKT; ++kt) {
    const int k0 = kt * KT;
    const bool more = (kt + 1) < NKT;

    // issue next-tile global loads early (double-buffer through registers)
    v4f nv0 = {}, nv1 = {};
    if (more) {
      const float* nsrc = arow_p + (k0 + KT);
      nv0 = *(const v4f*)(nsrc);
      nv1 = *(const v4f*)(nsrc + 4);
      if (kt + 2 < NKT) __builtin_prefetch(nsrc + KT, 0, 1);  // global_prefetch_b8
    }

    // ---- A fragments from current buffer (contiguous k order k = 16*lhalf + s, as for B) ----
    union F { v16h v; v8h h[2]; };
    F ahi[2], alo[2];
    const _Float16* base = &As[buf * BUFS];
#pragma unroll
    for (int m = 0; m < 2; ++m) {
      const int ab = (waveM * 32 + m * 16 + lrow) * LDA + lhalf * 16;
      ahi[m].h[0] = *(const v8h*)&base[ab];
      ahi[m].h[1] = *(const v8h*)&base[ab + 8];
      alo[m].h[0] = *(const v8h*)&base[ab + PL];
      alo[m].h[1] = *(const v8h*)&base[ab + PL + 8];
    }

    // ---- 24 WMMAs: B fragment reused across both M subtiles ----
#pragma unroll
    for (int s = 0; s < 4; ++s) {
      v16h bhi = *(const v16h*)(whi_l + (size_t)s * 16 * DO_ + k0);
      v16h blo = *(const v16h*)(wlo_l + (size_t)s * 16 * DO_ + k0);
#pragma unroll
      for (int m = 0; m < 2; ++m) {
        acc[m][s] = __builtin_amdgcn_wmma_f32_16x16x32_f16(
            false, alo[m].v, false, bhi, (short)0, acc[m][s], false, false);
        acc[m][s] = __builtin_amdgcn_wmma_f32_16x16x32_f16(
            false, ahi[m].v, false, blo, (short)0, acc[m][s], false, false);
        acc[m][s] = __builtin_amdgcn_wmma_f32_16x16x32_f16(
            false, ahi[m].v, false, bhi, (short)0, acc[m][s], false, false);
      }
    }

    // ---- stage next tile into the other buffer; single barrier ----
    if (more) store_strip(buf ^ 1, nv0, nv1);
    __syncthreads();
    buf ^= 1;
  }

  // ---- fused scatter-sum epilogue (C layout: VGPR r, lanes<16 -> M=r,
  //      lanes>=16 -> M=r+8; N = lane%16) ----
  float* out_b = out + (size_t)b * E_;
  const int nBase = waveN * 64 + lrow;
#pragma unroll
  for (int m = 0; m < 2; ++m) {
    const int mBase = n0 + waveM * 32 + m * 16 + lhalf * 8;
#pragma unroll
    for (int s = 0; s < 4; ++s) {
      const int xy = nBase + s * 16;
#pragma unroll
      for (int r = 0; r < 8; ++r) {
        const int e = idx[(mBase + r) * KHW + xy];
        atomicAdd(out_b + e, acc[m][s][r]);
      }
    }
  }
}

// ------------------------------ finalize -----------------------------------
// Analytic coverage count (kernel 16, stride 8, clamp to 511 => +8 at edge).
__device__ __forceinline__ int fd_cov(int h) {
  int nmin = (h >= 8) ? ((h - 8) >> 3) : 0;   // ceil((h-15)/8) clamped to 0
  int nmax = h >> 3;                          // <= 63 for h <= 511
  int c = nmax - nmin + 1;
  if (h == H_ - 1) c += 8;                    // clipped positions 512..519
  return c;
}

__global__ void fd_finalize(float* __restrict__ out) {
  size_t i = (size_t)blockIdx.x * blockDim.x + threadIdx.x;  // < B*E
  int e = (int)(i & (E_ - 1));
  int h = e >> 9;         // W_ = 512
  int w = e & (W_ - 1);
  float cnt = (float)(fd_cov(h) * fd_cov(w));
  out[i] = out[i] / cnt;
}

// ------------------------------- launcher ----------------------------------

extern "C" void kernel_launch(void* const* d_in, const int* in_sizes, int n_in,
                              void* d_out, int out_size, void* d_ws, size_t ws_size,
                              hipStream_t stream) {
  const float* tgt    = (const float*)d_in[0];   // [16, 4096, 768]
  const float* weight = (const float*)d_in[1];   // [16, 16, 768] -> [256,768]
  const int*   idx    = (const int*)d_in[2];     // [4096*256]
  float* out = (float*)d_out;                    // [16, 512, 512]

  _Float16* whi = (_Float16*)d_ws;                                   // 256*768 halves
  _Float16* wlo = (_Float16*)((char*)d_ws + (size_t)KHW * DO_ * 2);  // +384 KB

  // 1) zero the output accumulation buffer (16*E floats, float4 stores)
  fd_zero_out<<<(B_ * E_) / (256 * 4), 256, 0, stream>>>(out);

  // 2) split weight into f16 hi/lo planes
  fd_weight_split<<<(KHW * DO_) / 256, 256, 0, stream>>>(weight, whi, wlo);

  // 3) WMMA GEMM + fused atomic scatter-sum
  dim3 grid(N_TOK / BM, B_);                     // 64 x 16 workgroups
  fd_gemm_scatter<<<grid, 256, 0, stream>>>(tgt, whi, wlo, idx, out);

  // 4) divide by analytic coverage counts
  fd_finalize<<<(B_ * E_) / 256, 256, 0, stream>>>(out);
}
